// NeXtVLAD_27934467293513
// MI455X (gfx1250) — matrix operand order repaired and checked by reading, not compiled, not run on hardware
//
#include <hip/hip_runtime.h>

// ---------------------------------------------------------------------------
// NeXtVLAD forward on gfx1250: bf16 WMMA (v_wmma_f32_16x16x32_bf16) for all
// large GEMMs with 4-accumulator register blocking (A/B fragment reuse),
// wave32 VALU kernels for softmax/sigmoid/L1-norm glue.
// ---------------------------------------------------------------------------

typedef __attribute__((ext_vector_type(16))) __bf16 v16bf;
typedef __attribute__((ext_vector_type(8)))  float  v8f;

#define NN    64
#define MM    300
#define CC    768
#define EE    1536      // LAMB*C
#define GG    8
#define KK    64
#define GS    192       // EE/GG
#define DESC  12288     // KK*GS
#define NB    512
#define NCLS  701
#define ROWS  (NN*MM)   // 19200
#define PP    (MM*GG)   // 2400

// ---- helpers --------------------------------------------------------------
__device__ __forceinline__ unsigned short f2bf(float f) {
  unsigned int u = __float_as_uint(f);
  u += 0x7FFFu + ((u >> 16) & 1u);            // round to nearest even
  return (unsigned short)(u >> 16);
}
__device__ __forceinline__ float bf2f(unsigned short h) {
  return __uint_as_float(((unsigned int)h) << 16);
}
__device__ __forceinline__ float wave_sum32(float s) {
#pragma unroll
  for (int off = 16; off; off >>= 1) s += __shfl_xor(s, off, 32);
  return s;
}
__device__ __forceinline__ float wave_max32(float s) {
#pragma unroll
  for (int off = 16; off; off >>= 1) s = fmaxf(s, __shfl_xor(s, off, 32));
  return s;
}

// ---- conversion kernels ---------------------------------------------------
__global__ void k_f32_to_bf16(const float* __restrict__ src,
                              unsigned short* __restrict__ dst, size_t count) {
  size_t i = (size_t)blockIdx.x * blockDim.x + threadIdx.x;
  size_t stride = (size_t)gridDim.x * blockDim.x;
  for (; i < count; i += stride) dst[i] = f2bf(src[i]);
}

// features (N, M+1, C) -> x_bf (N*M, C), dropping row 0 of each sample
__global__ void k_conv_features(const float* __restrict__ feat,
                                unsigned short* __restrict__ xbf) {
  size_t i = (size_t)blockIdx.x * blockDim.x + threadIdx.x;
  if (i >= (size_t)ROWS * CC) return;
  size_t row = i / CC, c = i % CC;
  size_t n = row / MM, m = row % MM;
  xbf[i] = f2bf(feat[((n * (MM + 1)) + m + 1) * CC + c]);
}

// ---- generic WMMA GEMM: D(M,N) = A(M,K) * W(N,K)^T, bf16 in, f32 acc ------
// block = 128 (4 waves); wave w computes rows [bx*64 + w*16, +16) x
// cols [by*64, +64) with 4 accumulators; A fragment reused across 4 WMMAs.
// K multiple of 32, M multiple of 64.
template <bool NGUARD>
__global__ void k_gemm_nt(const unsigned short* __restrict__ A,
                          const unsigned short* __restrict__ W,
                          const float* __restrict__ bias,
                          const float* __restrict__ scale,
                          const float* __restrict__ shift,
                          float* __restrict__ Df,
                          unsigned short* __restrict__ Dbf,
                          int Ndim, int Kdim) {
  const int wave = threadIdx.x >> 5;
  const int lane = threadIdx.x & 31;
  const int row0 = blockIdx.x * 64 + wave * 16;
  const int colB = blockIdx.y * 64;
  const int ml = lane & 15;
  const int h  = lane >> 4;

  const unsigned short* Arow = A + (size_t)(row0 + ml) * Kdim;
  const unsigned short* Wrow[4];
#pragma unroll
  for (int c = 0; c < 4; ++c) {
    int wr = colB + c * 16 + ml;
    if (NGUARD && wr >= Ndim) wr = Ndim - 1;   // clamp (result discarded)
    Wrow[c] = W + (size_t)wr * Kdim;
  }

  v8f acc[4] = {};
  for (int kb = 0; kb < Kdim; kb += 32) {
    union { unsigned int u[8]; v16bf v; } af;
#pragma unroll
    for (int vi = 0; vi < 8; ++vi) {
      const int ka = h * 16 + vi * 2; // A 16x32 layout
      af.u[vi] = *(const unsigned int*)(Arow + kb + ka);
    }
    if (kb + 64 < Kdim) __builtin_prefetch(Arow + kb + 64, 0, 1);
#pragma unroll
    for (int c = 0; c < 4; ++c) {
      union { unsigned int u[8]; v16bf v; } bf;
#pragma unroll
      for (int vi = 0; vi < 8; ++vi)                           // B 32x16 layout
        bf.u[vi] = *(const unsigned int*)(Wrow[c] + kb + h * 16 + 2 * vi);
      acc[c] = __builtin_amdgcn_wmma_f32_16x16x32_bf16(
          false, af.v, false, bf.v, (short)0, acc[c], false, false);
    }
  }

#pragma unroll
  for (int c = 0; c < 4; ++c) {
    const int ncol = colB + c * 16 + ml;
    const bool nok = (!NGUARD) || (ncol < Ndim);
    float bia = 0.f, sc = 1.f, sh = 0.f;
    if (nok) {
      if (bias)  bia = bias[ncol];
      if (scale) { sc = scale[ncol]; sh = shift[ncol]; }
    }
#pragma unroll
    for (int r = 0; r < 8; ++r) {
      const int rr = row0 + r + h * 8;            // C/D f32 layout: M = r + h*8
      float val = (acc[c][r] + bia) * sc + sh;
      if (nok) {
        if (Df)  Df[(size_t)rr * Ndim + ncol] = val;
        if (Dbf) Dbf[(size_t)rr * Ndim + ncol] = f2bf(val);
      }
    }
  }
}

// ---- fcg: alpha_g = sigmoid(x_dot . fcg_w^T + fcg_b) ----------------------
// grid = ROWS, block = 256 (8 waves, one per group g)
__global__ void k_fcg_sigmoid(const unsigned short* __restrict__ xdot,
                              const unsigned short* __restrict__ fcgw,
                              const float* __restrict__ fcgb,
                              float* __restrict__ alphag) {
  const int row = blockIdx.x;
  const int g = threadIdx.x >> 5, lane = threadIdx.x & 31;
  const unsigned short* x = xdot + (size_t)row * EE;
  const unsigned short* w = fcgw + (size_t)g * EE;
  float s = 0.f;
  for (int e = lane; e < EE; e += 32) s += bf2f(x[e]) * bf2f(w[e]);
  s = wave_sum32(s);
  if (lane == 0)
    alphag[(size_t)row * GG + g] = 1.f / (1.f + __expf(-(s + fcgb[g])));
}

// ---- softmax(64) * sigmoid gate -> actT[n][k][p] (bf16) -------------------
// grid = ROWS, block = 256 (wave = group); lane covers k=lane and k=lane+32
__global__ void k_softmax_act(const float* __restrict__ wgkx,
                              const float* __restrict__ alphag,
                              const float* __restrict__ bn0g,
                              const float* __restrict__ bn0b,
                              unsigned short* __restrict__ actT) {
  const int row = blockIdx.x;                 // n*MM + m
  const int n = row / MM, m = row % MM;
  const int g = threadIdx.x >> 5, lane = threadIdx.x & 31;
  const float gsc = bn0g[m], gsh = bn0b[m];
  const float* base = wgkx + (size_t)row * (GG * KK) + g * KK;
  float v0 = base[lane] * gsc + gsh;
  float v1 = base[lane + 32] * gsc + gsh;
  float mx = wave_max32(fmaxf(v0, v1));
  float e0 = __expf(v0 - mx), e1 = __expf(v1 - mx);
  float s = wave_sum32(e0 + e1);
  const float r = alphag[(size_t)row * GG + g] / s;
  const size_t p = (size_t)m * GG + g;
  actT[((size_t)n * KK + lane)      * PP + p] = f2bf(e0 * r);
  actT[((size_t)n * KK + lane + 32) * PP + p] = f2bf(e1 * r);
}

// ---- a_sum[n*KK+k] = sum_p actT[n][k][p] ----------------------------------
__global__ void k_asum(const unsigned short* __restrict__ actT,
                       float* __restrict__ asum) {
  const int row = blockIdx.x * 8 + (threadIdx.x >> 5);  // 0..NN*KK-1
  const int lane = threadIdx.x & 31;
  const unsigned short* p = actT + (size_t)row * PP;
  float s = 0.f;
  for (int i = lane; i < PP; i += 32) s += bf2f(p[i]);
  s = wave_sum32(s);
  if (lane == 0) asum[row] = s;
}

// ---- batched VLAD GEMM: vg[n][k][d] = sum_p actT[n][k][p] * xtilde[n][p][d]
// A = actT (row-major 64x2400), B = xdot_bf[n] viewed as (2400x192) row-major.
// Each wave: one 16-wide d tile, ALL 64 k rows (4 accumulators); the strided
// B gather is done once per K-step and reused across 4 WMMAs.
// grid = (1, GS/64, NN), block = 128
__global__ void k_vlad_gemm(const unsigned short* __restrict__ actT,
                            const unsigned short* __restrict__ xdot,
                            float* __restrict__ vg) {
  const int n = blockIdx.z;
  const int wave = threadIdx.x >> 5, lane = threadIdx.x & 31;
  const int col0 = blockIdx.y * 64 + wave * 16;     // d
  const int ml = lane & 15, h = lane >> 4;

  const unsigned short* Abase = actT + (size_t)n * KK * PP;
  const unsigned short* B     = xdot + (size_t)n * MM * EE;  // (PP x GS) flat
  const int dcol = col0 + ml;

  v8f acc[4] = {};
  for (int kb = 0; kb < PP; kb += 32) {
    union { unsigned int u[8]; v16bf v; } bfrag;
#pragma unroll
    for (int vi = 0; vi < 8; ++vi) {
      const int p0 = kb + h * 16 + 2 * vi;          // strided gather for B
      unsigned int lo = B[(size_t)p0 * GS + dcol];
      unsigned int hi = B[(size_t)(p0 + 1) * GS + dcol];
      bfrag.u[vi] = lo | (hi << 16);
    }
#pragma unroll
    for (int t = 0; t < 4; ++t) {
      union { unsigned int u[8]; v16bf v; } af;
      const unsigned short* Arow = Abase + (size_t)(t * 16 + ml) * PP;
#pragma unroll
      for (int vi = 0; vi < 8; ++vi) {
        const int ka = h * 16 + vi * 2;
        af.u[vi] = *(const unsigned int*)(Arow + kb + ka);
      }
      acc[t] = __builtin_amdgcn_wmma_f32_16x16x32_bf16(
          false, af.v, false, bfrag.v, (short)0, acc[t], false, false);
    }
  }
#pragma unroll
  for (int t = 0; t < 4; ++t) {
#pragma unroll
    for (int r = 0; r < 8; ++r) {
      const int k = t * 16 + r + h * 8;
      vg[((size_t)n * KK + k) * GS + dcol] = acc[t][r];
    }
  }
}

// ---- L1 norm over d, scale, write bf16 vlad at [n][d*64+k] ----------------
__global__ void k_vlad_norm(const float* __restrict__ vg,
                            const float* __restrict__ asum,
                            const float* __restrict__ cw2,
                            const float* __restrict__ bn1g,
                            const float* __restrict__ bn1b,
                            unsigned short* __restrict__ vladbf) {
  const int row = blockIdx.x * 8 + (threadIdx.x >> 5);  // n*KK + k
  const int lane = threadIdx.x & 31;
  const int n = row >> 6, k = row & 63;
  const float* src = vg + (size_t)row * GS;
  const float av = asum[row];
  float vals[6], s = 0.f;
#pragma unroll
  for (int i = 0; i < 6; ++i) {
    const int d = lane + 32 * i;
    float v = src[d] - av * cw2[(size_t)d * KK + k];
    vals[i] = v;
    s += fabsf(v);
  }
  s = fmaxf(wave_sum32(s), 1e-12f);
  const float g = bn1g[0], b = bn1b[0];
#pragma unroll
  for (int i = 0; i < 6; ++i) {
    const int d = lane + 32 * i;
    vladbf[(size_t)n * DESC + (size_t)d * KK + k] = f2bf(vals[i] / s * g + b);
  }
}

// ---------------------------------------------------------------------------
extern "C" void kernel_launch(void* const* d_in, const int* in_sizes, int n_in,
                              void* d_out, int out_size, void* d_ws, size_t ws_size,
                              hipStream_t stream) {
  (void)in_sizes; (void)n_in; (void)out_size; (void)ws_size;
  const float* features = (const float*)d_in[0];
  const float* fc0_w    = (const float*)d_in[1];
  const float* fc0_b    = (const float*)d_in[2];
  const float* fcgk_w   = (const float*)d_in[3];
  const float* fcgk_b   = (const float*)d_in[4];
  const float* fcg_w    = (const float*)d_in[5];
  const float* fcg_b    = (const float*)d_in[6];
  const float* bn0_g    = (const float*)d_in[7];
  const float* bn0_b    = (const float*)d_in[8];
  const float* cw2      = (const float*)d_in[9];
  const float* bn1_g    = (const float*)d_in[10];
  const float* bn1_b    = (const float*)d_in[11];
  const float* cls_fc_w = (const float*)d_in[12];
  const float* cls_fc_b = (const float*)d_in[13];
  const float* cls_bn_g = (const float*)d_in[14];
  const float* cls_bn_b = (const float*)d_in[15];
  const float* cls_w    = (const float*)d_in[16];
  const float* cls_b    = (const float*)d_in[17];

  // workspace layout (bytes, 256-aligned)
  char* ws = (char*)d_ws;
  size_t off = 0;
  auto alloc = [&](size_t bytes) {
    char* p = ws + off;
    off += (bytes + 255) & ~(size_t)255;
    return p;
  };
  unsigned short* x_bf     = (unsigned short*)alloc((size_t)ROWS * CC * 2);
  unsigned short* fc0w_bf  = (unsigned short*)alloc((size_t)EE * CC * 2);
  unsigned short* xdot_bf  = (unsigned short*)alloc((size_t)ROWS * EE * 2);
  unsigned short* fcgkw_bf = (unsigned short*)alloc((size_t)(GG * KK) * EE * 2);
  unsigned short* fcgw_bf  = (unsigned short*)alloc((size_t)GG * EE * 2);
  float*          wgkx     = (float*)alloc((size_t)ROWS * GG * KK * 4);
  float*          alphag   = (float*)alloc((size_t)ROWS * GG * 4);
  unsigned short* actT     = (unsigned short*)alloc((size_t)NN * KK * PP * 2);
  float*          asum     = (float*)alloc((size_t)NN * KK * 4);
  float*          vg       = (float*)alloc((size_t)NN * KK * GS * 4);
  unsigned short* vlad_bf  = (unsigned short*)alloc((size_t)NN * DESC * 2);
  unsigned short* clsfw_bf = (unsigned short*)alloc((size_t)NB * DESC * 2);
  unsigned short* f_bf     = (unsigned short*)alloc((size_t)NN * NB * 2);
  unsigned short* clsw_bf  = (unsigned short*)alloc((size_t)NCLS * NB * 2);

  float* cls_out = (float*)d_out;                      // (NN, NCLS)
  float* f_out   = (float*)d_out + (size_t)NN * NCLS;  // (NN, NB)

  // 1) conversions
  {
    size_t cnt = (size_t)ROWS * CC;
    k_conv_features<<<(unsigned)((cnt + 255) / 256), 256, 0, stream>>>(features, x_bf);
    k_f32_to_bf16<<<2048, 256, 0, stream>>>(fc0_w, fc0w_bf, (size_t)EE * CC);
    k_f32_to_bf16<<<2048, 256, 0, stream>>>(fcgk_w, fcgkw_bf, (size_t)GG * KK * EE);
    k_f32_to_bf16<<<64,   256, 0, stream>>>(fcg_w, fcgw_bf, (size_t)GG * EE);
    k_f32_to_bf16<<<4096, 256, 0, stream>>>(cls_fc_w, clsfw_bf, (size_t)NB * DESC);
    k_f32_to_bf16<<<1024, 256, 0, stream>>>(cls_w, clsw_bf, (size_t)NCLS * NB);
  }

  // 2) GEMM1: x_dot = x @ fc0_w^T + fc0_b   (bf16 out only)
  k_gemm_nt<false><<<dim3(ROWS / 64, EE / 64), 128, 0, stream>>>(
      x_bf, fc0w_bf, fc0_b, nullptr, nullptr, nullptr, xdot_bf, EE, CC);

  // 3) GEMM2: WgkX = x_dot @ fcgk_w^T + fcgk_b   (f32 out)
  k_gemm_nt<false><<<dim3(ROWS / 64, (GG * KK) / 64), 128, 0, stream>>>(
      xdot_bf, fcgkw_bf, fcgk_b, nullptr, nullptr, wgkx, nullptr, GG * KK, EE);

  // 4) alpha_g = sigmoid(x_dot @ fcg_w^T + fcg_b)
  k_fcg_sigmoid<<<ROWS, 256, 0, stream>>>(xdot_bf, fcgw_bf, fcg_b, alphag);

  // 5) activation = softmax(bn0(WgkX)) * alpha_g  -> actT[n][k][p]
  k_softmax_act<<<ROWS, 256, 0, stream>>>(wgkx, alphag, bn0_g, bn0_b, actT);

  // 6) a_sum[n,k] = sum_p activation
  k_asum<<<(NN * KK) / 8, 256, 0, stream>>>(actT, asum);

  // 7) vg[n][k][d] = sum_p act * x_tilde
  k_vlad_gemm<<<dim3(1, GS / 64, NN), 128, 0, stream>>>(actT, xdot_bf, vg);

  // 8) subtract a, L1 normalize over d, bn1 -> vlad bf16 (desc = d*64+k)
  k_vlad_norm<<<(NN * KK) / 8, 256, 0, stream>>>(vg, asum, cw2, bn1_g, bn1_b, vlad_bf);

  // 9) f = bn(vlad @ cls_fc_w^T + cls_fc_b)  -> f32 output + bf16 copy
  k_gemm_nt<false><<<dim3(NN / 64, NB / 64), 128, 0, stream>>>(
      vlad_bf, clsfw_bf, cls_fc_b, cls_bn_g, cls_bn_b, f_out, f_bf, NB, DESC);

  // 10) cls = f @ cls_w^T + cls_b   (N=701 guarded)
  k_gemm_nt<true><<<dim3(NN / 64, (NCLS + 63) / 64), 128, 0, stream>>>(
      f_bf, clsw_bf, cls_b, nullptr, nullptr, cls_out, nullptr, NCLS, NB);
}
